// PCAM_49323404427660
// MI455X (gfx1250) — matrix operand order repaired and checked by reading, not compiled, not run on hardware
//
#include <hip/hip_runtime.h>
#include <hip/hip_bf16.h>

// ---------------- constants ----------------
#define BATCH 4
#define NPIX  4096           // H*W
#define CCH   64
#define NC    (NPIX * CCH)   // 262144 elements per batch
#define MTOT  (BATCH * NPIX) // 16384 rows for projection GEMM

typedef _Float16 h16 __attribute__((ext_vector_type(16)));
typedef float    f8  __attribute__((ext_vector_type(8)));

// ---------------- WMMA wrapper ----------------
__device__ __forceinline__ f8 wmma_f16(h16 a, h16 b, f8 c) {
    return __builtin_amdgcn_wmma_f32_16x16x32_f16(
        /*neg_a=*/false, a, /*neg_b=*/false, b,
        /*c_mod=*/(short)0, c, /*reuse_a=*/false, /*reuse_b=*/false);
}

// ---------------- fragment loaders ----------------
// A-matrix 16x32 f16 fragment, row-major source (ld in elements).
// ISA layout: lanes 0-15 row M=lane, elems: K=0..7 then 16..23;
//             lanes 16-31 same rows, K=8..15 then 24..31.
__device__ __forceinline__ h16 load_frag_a(const _Float16* p, int ld, int lane) {
    int row = lane & 15;
    int hf  = (lane >> 4) & 1;
    const _Float16* r = p + (size_t)row * ld + hf * 8;
    h16 v;
#pragma unroll
    for (int e = 0; e < 8; ++e) v[e]     = r[e];
#pragma unroll
    for (int e = 0; e < 8; ++e) v[8 + e] = r[16 + e];
    return v;
}

// Same as load_frag_a but converts from f32 source.
__device__ __forceinline__ h16 load_frag_a_f32(const float* p, int ld, int lane) {
    int row = lane & 15;
    int hf  = (lane >> 4) & 1;
    const float* r = p + (size_t)row * ld + hf * 8;
    h16 v;
#pragma unroll
    for (int e = 0; e < 8; ++e) v[e]     = (_Float16)r[e];
#pragma unroll
    for (int e = 0; e < 8; ++e) v[8 + e] = (_Float16)r[16 + e];
    return v;
}

// B-matrix 32x16 fragment where B[k][n] = p[n*ld + k] (i.e. the source holds
// the *transpose* of B row-major). ISA B layout: lane holds column n=lane&15,
// elems: K=8*(lane>>4)+0..7 then 16+8*(lane>>4)+0..7 (matches A fragment).
__device__ __forceinline__ h16 load_frag_bT(const _Float16* p, int ld, int lane) {
    int n  = lane & 15;
    int hf = (lane >> 4) & 1;
    const _Float16* r = p + (size_t)n * ld + hf * 8;
    h16 v;
#pragma unroll
    for (int e = 0; e < 8; ++e) v[e]     = r[e];
#pragma unroll
    for (int e = 0; e < 8; ++e) v[8 + e] = r[16 + e];
    return v;
}

// ---------------- CDNA5 helpers ----------------
__device__ __forceinline__ unsigned lds_off(const void* p) {
    // generic LDS address: low 32 bits are the LDS byte offset
    return (unsigned)(uintptr_t)p;
}

// async copy 16B/lane from global to LDS (tracked by ASYNCcnt)
__device__ __forceinline__ void async_copy_b128(unsigned lds_addr, unsigned voff,
                                                unsigned long long gbase) {
    asm volatile("global_load_async_to_lds_b128 %0, %1, %2"
                 :: "v"(lds_addr), "v"(voff), "s"(gbase)
                 : "memory");
}
__device__ __forceinline__ void wait_async0() {
    asm volatile("s_wait_asynccnt 0x0" ::: "memory");
}
__device__ __forceinline__ void wait_ds0() {
    asm volatile("s_wait_dscnt 0x0" ::: "memory");
}

// ================= kernel 1: six 1x1-conv projections =================
struct ProjArgs {
    const float* w[6];
    const float* bias[6];
};

__global__ __launch_bounds__(128)
void proj_kernel(const float* __restrict__ x, ProjArgs pa, _Float16* __restrict__ outbase) {
    __shared__ _Float16 wt[CCH * CCH];   // W transposed: wt[d*64 + c] = W[c][d]
    __shared__ float    blds[CCH];

    int pj = blockIdx.y;
    const float* W    = pa.w[pj];
    const float* bias = pa.bias[pj];
    _Float16* out = outbase + (size_t)pj * MTOT * CCH;

    for (int i = threadIdx.x; i < CCH * CCH; i += 128) {
        int c = i >> 6, d = i & 63;                  // reads coalesced in d
        wt[d * CCH + c] = (_Float16)W[i];
    }
    for (int i = threadIdx.x; i < CCH; i += 128) blds[i] = bias[i];
    __syncthreads();

    int lane = threadIdx.x & 31;
    int wv   = threadIdx.x >> 5;
    int row0 = blockIdx.x * 64 + wv * 16;

    h16 a0 = load_frag_a_f32(x + (size_t)row0 * CCH + 0,  CCH, lane);
    h16 a1 = load_frag_a_f32(x + (size_t)row0 * CCH + 32, CCH, lane);

    int nn = lane & 15, hf = lane >> 4;
#pragma unroll
    for (int nt = 0; nt < 4; ++nt) {
        f8 acc = {};
        // B[c][d]=W[c][d]=wt[d*64+c]: contiguous fragment reads
        h16 b0 = load_frag_bT(wt + (size_t)(nt * 16) * CCH + 0,  CCH, lane);
        h16 b1 = load_frag_bT(wt + (size_t)(nt * 16) * CCH + 32, CCH, lane);
        acc = wmma_f16(a0, b0, acc);
        acc = wmma_f16(a1, b1, acc);
        float bv = blds[nt * 16 + nn];
#pragma unroll
        for (int g = 0; g < 8; ++g) {
            int m = g + 8 * hf;
            out[(size_t)(row0 + m) * CCH + nt * 16 + nn] = (_Float16)(acc[g] + bv);
        }
    }
}

// ================= kernel 2: spatial flash attention =================
__global__ __launch_bounds__(128)
void spat_attn_kernel(const _Float16* __restrict__ Qs, const _Float16* __restrict__ Ks,
                      const _Float16* __restrict__ Vs, float* __restrict__ spat) {
    __shared__ _Float16 klds[32 * CCH];          // K chunk, row-major [key][ch]
    __shared__ _Float16 vt[CCH * 32];            // V chunk transposed [ch][key]
    __shared__ _Float16 plds[4][16 * 32];        // per-wave P tiles

    int b    = blockIdx.y;
    int lane = threadIdx.x & 31;
    int wv   = threadIdx.x >> 5;
    int q0   = blockIdx.x * 64 + wv * 16;

    const _Float16* Qb = Qs + (size_t)b * NC;
    const _Float16* Kb = Ks + (size_t)b * NC;
    const _Float16* Vb = Vs + (size_t)b * NC;

    h16 qa0 = load_frag_a(Qb + (size_t)q0 * CCH + 0,  CCH, lane);
    h16 qa1 = load_frag_a(Qb + (size_t)q0 * CCH + 32, CCH, lane);

    f8 o[4];
#pragma unroll
    for (int t = 0; t < 4; ++t) o[t] = (f8){};
    float m_i[8], l_i[8];
#pragma unroll
    for (int g = 0; g < 8; ++g) { m_i[g] = -INFINITY; l_i[g] = 0.f; }

    int nn = lane & 15, hf = lane >> 4;
    unsigned klds_base = lds_off(klds);

    for (int kc = 0; kc < NPIX / 32; ++kc) {
        __syncthreads();  // previous chunk's LDS reads complete

        // ---- K chunk: async DMA global -> LDS (4 KB, 2 x b128 per lane) ----
        {
            unsigned long long gk = (unsigned long long)(Kb + (size_t)kc * 32 * CCH);
#pragma unroll
            for (int i = 0; i < 2; ++i) {
                unsigned byte_off = threadIdx.x * 16u + i * 2048u;
                async_copy_b128(klds_base + byte_off, byte_off, gk);
            }
        }
        // ---- V chunk: coalesced b128 reads, transposed scalar LDS stores ----
        {
            int key = threadIdx.x >> 2;
            int ch0 = (threadIdx.x & 3) * 16;
            const _Float16* src = Vb + ((size_t)kc * 32 + key) * CCH + ch0;
            union { uint4 q; _Float16 h[8]; } u0, u1;
            u0.q = *(const uint4*)(src);
            u1.q = *(const uint4*)(src + 8);
#pragma unroll
            for (int e = 0; e < 8; ++e) vt[(ch0 + e) * 32 + key]     = u0.h[e];
#pragma unroll
            for (int e = 0; e < 8; ++e) vt[(ch0 + 8 + e) * 32 + key] = u1.h[e];
        }
        wait_async0();      // ASYNCcnt not covered by the barrier's waits
        __syncthreads();

        // ---- S = Q · K^T (two 16x16 key sub-tiles, K-dim = 64 channels) ----
        f8 s0 = {}, s1 = {};
        s0 = wmma_f16(qa0, load_frag_bT(klds + 0,             CCH, lane), s0);
        s0 = wmma_f16(qa1, load_frag_bT(klds + 32,            CCH, lane), s0);
        s1 = wmma_f16(qa0, load_frag_bT(klds + 16 * CCH + 0,  CCH, lane), s1);
        s1 = wmma_f16(qa1, load_frag_bT(klds + 16 * CCH + 32, CCH, lane), s1);

        // ---- online softmax (rows live in VGPR index g + half-wave) ----
        float alpha[8];
#pragma unroll
        for (int g = 0; g < 8; ++g) {
            float v0 = s0[g] * 0.125f, v1 = s1[g] * 0.125f;
            float mx = fmaxf(v0, v1);
#pragma unroll
            for (int off = 8; off >= 1; off >>= 1)
                mx = fmaxf(mx, __shfl_xor(mx, off, 32));
            float mn = fmaxf(m_i[g], mx);
            alpha[g] = __expf(m_i[g] - mn);
            float p0 = __expf(v0 - mn);
            float p1 = __expf(v1 - mn);
            s0[g] = p0; s1[g] = p1;
            float rs = p0 + p1;
#pragma unroll
            for (int off = 8; off >= 1; off >>= 1)
                rs += __shfl_xor(rs, off, 32);
            l_i[g] = l_i[g] * alpha[g] + rs;
            m_i[g] = mn;
        }
#pragma unroll
        for (int t = 0; t < 4; ++t)
#pragma unroll
            for (int g = 0; g < 8; ++g) o[t][g] *= alpha[g];

        // ---- P: D-layout -> LDS -> A-layout (per-wave private region) ----
        _Float16* pl = plds[wv];
#pragma unroll
        for (int g = 0; g < 8; ++g) {
            int m = g + 8 * hf;
            pl[m * 32 + nn]      = (_Float16)s0[g];
            pl[m * 32 + nn + 16] = (_Float16)s1[g];
        }
        wait_ds0();          // intra-wave DS ordering, no cross-wave barrier needed
        h16 pfa = load_frag_a(pl, 32, lane);

        // ---- O += P · V : B[k][n]=V[k][n]=vt[n*32+k], contiguous frags ----
#pragma unroll
        for (int t = 0; t < 4; ++t)
            o[t] = wmma_f16(pfa, load_frag_bT(vt + (size_t)(t * 16) * 32, 32, lane), o[t]);
    }

    // ---- normalize and store ----
    float* outb = spat + (size_t)b * NC;
#pragma unroll
    for (int t = 0; t < 4; ++t) {
#pragma unroll
        for (int g = 0; g < 8; ++g) {
            int m = g + 8 * hf;
            outb[(size_t)(q0 + m) * CCH + t * 16 + nn] = o[t][g] / l_i[g];
        }
    }
}

// ================= kernel 3a: channel Gram + softmax =================
__global__ __launch_bounds__(512)
void chan_gram_kernel(const _Float16* __restrict__ Qc, const _Float16* __restrict__ Kc,
                      _Float16* __restrict__ attc) {
    __shared__ float glds[CCH * CCH];
    int b = blockIdx.x;
    const _Float16* Qb = Qc + (size_t)b * NC;
    const _Float16* Kb = Kc + (size_t)b * NC;
    int lane = threadIdx.x & 31;
    int wv   = threadIdx.x >> 5;       // 16 waves -> 4x4 tile grid
    int tr = wv >> 2, tc = wv & 3;

    f8 acc = {};
    for (int k0 = 0; k0 < NPIX; k0 += 32) {
        h16 a  = load_frag_a (Qb + (size_t)(tr * 16) * NPIX + k0, NPIX, lane);
        h16 bb = load_frag_bT(Kb + (size_t)(tc * 16) * NPIX + k0, NPIX, lane);
        acc = wmma_f16(a, bb, acc);
    }
    int nn = lane & 15, hf = lane >> 4;
#pragma unroll
    for (int g = 0; g < 8; ++g)
        glds[(tr * 16 + g + 8 * hf) * CCH + tc * 16 + nn] = acc[g] * 0.125f;
    __syncthreads();

    if (threadIdx.x < CCH) {  // one thread per row of the 64x64 Gram
        float* r = glds + threadIdx.x * CCH;
        float mx = -INFINITY;
        for (int j = 0; j < CCH; ++j) mx = fmaxf(mx, r[j]);
        float s = 0.f;
        for (int j = 0; j < CCH; ++j) { float e = __expf(r[j] - mx); r[j] = e; s += e; }
        float inv = 1.f / s;
        _Float16* ab = attc + (size_t)b * CCH * CCH;
        for (int j = 0; j < CCH; ++j) ab[threadIdx.x * CCH + j] = (_Float16)(r[j] * inv);
    }
}

// ================= kernel 3b: chan = att_c @ Vc =================
__global__ __launch_bounds__(128)
void chan_apply_kernel(const _Float16* __restrict__ attc, const _Float16* __restrict__ Vc,
                       float* __restrict__ chan) {
    __shared__ _Float16 vt[16 * CCH];  // V tile transposed: vt[col*64 + k]
    int b    = blockIdx.y;
    int tcb  = blockIdx.x;             // 256 column tiles of 16 along N=4096
    int lane = threadIdx.x & 31;
    int tr   = threadIdx.x >> 5;       // 4 row tiles of 16 along C=64

    const _Float16* A  = attc + (size_t)b * CCH * CCH;
    const _Float16* Vb = Vc   + (size_t)b * NC;

    // stage 64x16 V tile transposed (coalesced b128 reads, scalar LDS stores)
    {
        int k  = threadIdx.x >> 1;
        int pt = threadIdx.x & 1;
        const _Float16* src = Vb + (size_t)k * NPIX + tcb * 16 + pt * 8;
        union { uint4 q; _Float16 h[8]; } u;
        u.q = *(const uint4*)src;
#pragma unroll
        for (int e = 0; e < 8; ++e) vt[(pt * 8 + e) * CCH + k] = u.h[e];
    }
    __syncthreads();

    f8 acc = {};
#pragma unroll
    for (int ks = 0; ks < 2; ++ks) {
        h16 a  = load_frag_a (A + (size_t)(tr * 16) * CCH + ks * 32, CCH, lane);
        h16 bb = load_frag_bT(vt + ks * 32, CCH, lane);
        acc = wmma_f16(a, bb, acc);
    }
    float* cb = chan + (size_t)b * NC;
    int nn = lane & 15, hf = lane >> 4;
#pragma unroll
    for (int g = 0; g < 8; ++g)
        cb[(size_t)(tr * 16 + g + 8 * hf) * NPIX + tcb * 16 + nn] = acc[g];
}

// ================= kernel 4: fusion =================
__global__ __launch_bounds__(256)
void fuse_kernel(const float* __restrict__ x, const float* __restrict__ spat,
                 const float* __restrict__ chan, const float* __restrict__ a_s,
                 const float* __restrict__ b_s, float* __restrict__ out) {
    size_t i = (size_t)blockIdx.x * blockDim.x + threadIdx.x;
    float av = a_s[0], bv = b_s[0];
    out[i] = bv * spat[i] + av * chan[i] + 2.f * x[i];
}

// ================= host launcher =================
extern "C" void kernel_launch(void* const* d_in, const int* in_sizes, int n_in,
                              void* d_out, int out_size, void* d_ws, size_t ws_size,
                              hipStream_t stream) {
    (void)in_sizes; (void)n_in; (void)out_size; (void)ws_size;

    const float* x = (const float*)d_in[0];
    ProjArgs pa;
    // order: Qs, Ks, Vs, Qc, Kc, Vc
    pa.w[0] = (const float*)d_in[1];  pa.bias[0] = (const float*)d_in[2];
    pa.w[1] = (const float*)d_in[3];  pa.bias[1] = (const float*)d_in[4];
    pa.w[2] = (const float*)d_in[5];  pa.bias[2] = (const float*)d_in[6];
    pa.w[3] = (const float*)d_in[7];  pa.bias[3] = (const float*)d_in[8];
    pa.w[4] = (const float*)d_in[9];  pa.bias[4] = (const float*)d_in[10];
    pa.w[5] = (const float*)d_in[11]; pa.bias[5] = (const float*)d_in[12];
    const float* a_s = (const float*)d_in[13];
    const float* b_s = (const float*)d_in[14];

    // workspace layout
    char* ws = (char*)d_ws;
    _Float16* proj = (_Float16*)ws;                       // 6 x 2 MB (f16)
    _Float16* Qs = proj + 0 * (size_t)MTOT * CCH;
    _Float16* Ks = proj + 1 * (size_t)MTOT * CCH;
    _Float16* Vs = proj + 2 * (size_t)MTOT * CCH;
    _Float16* Qc = proj + 3 * (size_t)MTOT * CCH;
    _Float16* Kc = proj + 4 * (size_t)MTOT * CCH;
    _Float16* Vc = proj + 5 * (size_t)MTOT * CCH;
    float* spat  = (float*)(ws + 6 * (size_t)MTOT * CCH * 2);            // 4 MB
    float* chan  = (float*)(ws + 6 * (size_t)MTOT * CCH * 2 + 4194304);  // 4 MB
    _Float16* attc = (_Float16*)(ws + 6 * (size_t)MTOT * CCH * 2 + 8388608); // 32 KB

    // 1) projections: grid (256 row-blocks, 6 matrices)
    proj_kernel<<<dim3(MTOT / 64, 6), 128, 0, stream>>>(x, pa, proj);

    // 2) spatial flash attention: grid (64 q-superblocks, B)
    spat_attn_kernel<<<dim3(NPIX / 64, BATCH), 128, 0, stream>>>(Qs, Ks, Vs, spat);

    // 3) channel attention
    chan_gram_kernel<<<dim3(BATCH), 512, 0, stream>>>(Qc, Kc, attc);
    chan_apply_kernel<<<dim3(NPIX / 16, BATCH), 128, 0, stream>>>(attc, Vc, chan);

    // 4) fusion
    fuse_kernel<<<dim3((BATCH * NC) / 256), 256, 0, stream>>>(x, spat, chan, a_s, b_s,
                                                              (float*)d_out);
}
